// DynamicUNet_74594991997047
// MI455X (gfx1250) — matrix operand order repaired and checked by reading, not compiled, not run on hardware
//
#include <hip/hip_runtime.h>

// ---------------------------------------------------------------------------
// CDNA5 (gfx1250) dynamic-UNet. Channel GEMMs: bf16 operands staged through
// LDS, v_wmma_f32_16x16x32_bf16 compute, f32 accumulate.
// ---------------------------------------------------------------------------

typedef __attribute__((ext_vector_type(16))) __bf16        v16bf;
typedef __attribute__((ext_vector_type(8)))  float         v8f;
typedef __attribute__((ext_vector_type(8)))  unsigned int  v8u;

#define B_N 2
#define KP  40   // padded LDS K-stride (halves): 80B rows -> 16B aligned, bank-spread

__device__ __forceinline__ unsigned short f2bfu(float f) {
  unsigned int u = __builtin_bit_cast(unsigned int, f);
  u += 0x7FFFu + ((u >> 16) & 1u);   // round-to-nearest-even
  return (unsigned short)(u >> 16);
}

// ---------------------------------------------------------------------------
// Fused 1x1-conv -> ReLU -> spatial mean:  M[b,j] = mean_hw relu(b_j + sum_c W[j,c]*X[b,c,hw])
// ---------------------------------------------------------------------------
__global__ void meanrelu_gen_kernel(const float* __restrict__ X, const float* __restrict__ W,
                                    const float* __restrict__ bias, float* __restrict__ M,
                                    int C, int HW, int R) {
  __shared__ float w_s[1024];
  __shared__ float red[256];
  const int bj = blockIdx.x;
  const int b = bj / R, j = bj % R;
  const int tid = threadIdx.x;
  for (int c = tid; c < C; c += blockDim.x) w_s[c] = W[(size_t)j * C + c];
  __syncthreads();
  const float bjv = bias[j];
  const float* Xb = X + (size_t)b * C * HW;
  float acc = 0.f;
  for (int hw = tid; hw < HW; hw += blockDim.x) {
    float s = bjv;
    for (int c = 0; c < C; ++c) s += w_s[c] * Xb[(size_t)c * HW + hw];
    acc += fmaxf(s, 0.f);
  }
  red[tid] = acc;
  __syncthreads();
  for (int st = 128; st > 0; st >>= 1) {
    if (tid < st) red[tid] += red[tid + st];
    __syncthreads();
  }
  if (tid == 0) M[bj] = red[0] / (float)HW;
}

// ---------------------------------------------------------------------------
// Tiny FC (f32 out): Out[b,i] = bias[i] + sum_j M[b,j]*W[i,j]   (W row-major [N,R])
// ---------------------------------------------------------------------------
__global__ void fc_gen_kernel(const float* __restrict__ M, const float* __restrict__ W,
                              const float* __restrict__ bias, float* __restrict__ Out,
                              int R, int N) {
  const int idx = blockIdx.x * blockDim.x + threadIdx.x;
  if (idx >= N) return;
  const int b = blockIdx.y;
  const float* m = M + (size_t)b * R;
  const float* w = W + (size_t)idx * R;
  float s = bias[idx];
  for (int r = 0; r < R; ++r) s += m[r] * w[r];
  Out[(size_t)b * N + idx] = s;
}

// ---------------------------------------------------------------------------
// Tiny FC -> bf16, padded weight layout: Out[b, o*Cpad + c] for c<Cin is row
// (o*Cin + c) of the generator output; zero for c in [Cin, Cpad).
// ---------------------------------------------------------------------------
__global__ void fc_gen_bf16_kernel(const float* __restrict__ M, const float* __restrict__ W,
                                   const float* __restrict__ bias, unsigned short* __restrict__ Out,
                                   int R, int Cin, int Cpad, int Ntot) {
  const int idx = blockIdx.x * blockDim.x + threadIdx.x;
  if (idx >= Ntot) return;
  const int b = blockIdx.y;
  const int o = idx / Cpad;
  const int c = idx % Cpad;
  float s = 0.f;
  if (c < Cin) {
    const int row = o * Cin + c;
    const float* m = M + (size_t)b * R;
    const float* w = W + (size_t)row * R;
    s = bias[row];
    for (int r = 0; r < R; ++r) s += m[r] * w[r];
  }
  Out[(size_t)b * Ntot + idx] = f2bfu(s);
}

// ---------------------------------------------------------------------------
// f32 -> bf16 conversion (weights).
// ---------------------------------------------------------------------------
__global__ void cvt_bf16_kernel(const float* __restrict__ in, unsigned short* __restrict__ out, int n) {
  const int i = blockIdx.x * blockDim.x + threadIdx.x;
  if (i < n) out[i] = f2bfu(in[i]);
}

__global__ void zero_bf16_kernel(unsigned short* __restrict__ p, int n) {
  const int i = blockIdx.x * blockDim.x + threadIdx.x;
  if (i < n) p[i] = 0;
}

// ---------------------------------------------------------------------------
// Dynamic depthwise 3x3, pad 1; bf16 output with padded channel dim Cpad.
// ---------------------------------------------------------------------------
__global__ void dw3x3_kernel(const float* __restrict__ X, const float* __restrict__ Kd,
                             unsigned short* __restrict__ Y, int C, int Cpad, int H, int W) {
  const int bc = blockIdx.y;
  const int b = bc / C, c = bc % C;
  const float* kp = Kd + (size_t)bc * 9;
  float k[9];
#pragma unroll
  for (int i = 0; i < 9; ++i) k[i] = kp[i];
  const float* xp = X + (size_t)bc * H * W;
  unsigned short* yp = Y + ((size_t)b * Cpad + c) * (size_t)(H * W);
  const int HW = H * W;
  for (int hw = blockIdx.x * blockDim.x + threadIdx.x; hw < HW; hw += gridDim.x * blockDim.x) {
    const int h = hw / W, w = hw % W;
    float s = 0.f;
#pragma unroll
    for (int dy = 0; dy < 3; ++dy) {
      const int hh = h + dy - 1;
      if (hh < 0 || hh >= H) continue;
#pragma unroll
      for (int dx = 0; dx < 3; ++dx) {
        const int ww = w + dx - 1;
        if (ww < 0 || ww >= W) continue;
        s += k[dy * 3 + dx] * xp[hh * W + ww];
      }
    }
    yp[hw] = f2bfu(s);
  }
}

// ---------------------------------------------------------------------------
// WMMA GEMM (bf16 x bf16 -> f32):
//   Y[b,o,s] = act(bias[o] + sum_c Wm[b*wStrideB + o*C + c] * X[b*C*S + c*S + s])
// Workgroup: 256 thr = 8 waves; tile 32 O x 256 S; K-step 32.
// LDS: X tile stored transposed (XsT[s][k], stride KP) -> per-lane B fragment
// is two aligned 16B ds loads; W tile (Ws[o][k], stride KP) -> A fragment is
// two aligned 16B ds loads. Requires C%32==0, O%32==0, S%256==0 (all call sites).
// Fragment layouts per CDNA5 ISA 7.12.2.
// ---------------------------------------------------------------------------
template <bool OUTBF>
__global__ void wmma_gemm_kernel(const unsigned short* __restrict__ X,
                                 const unsigned short* __restrict__ Wm,
                                 const float* __restrict__ bias, void* __restrict__ Yv,
                                 int C, int O, int S, long wStrideB, int relu) {
  __shared__ __align__(16) unsigned short XsT[256 * KP];  // 20 KB
  __shared__ __align__(16) unsigned short Ws[32 * KP];    // 2.5 KB

  const int tid   = threadIdx.x;
  const int lane  = tid & 31;
  const int wave  = tid >> 5;       // 0..7
  const int waveO = wave >> 2;      // 0..1
  const int waveS = wave & 3;       // 0..3
  const int b     = blockIdx.z;
  const int oB0   = blockIdx.y * 32;
  const int sB0   = blockIdx.x * 256;
  const int half  = lane >> 4;      // 0 or 1
  const int l16   = lane & 15;

  const unsigned short* Wb = Wm + (size_t)b * (size_t)wStrideB;
  const unsigned short* Xb = X + (size_t)b * C * S;

  const v8f zero = {0.f, 0.f, 0.f, 0.f, 0.f, 0.f, 0.f, 0.f};
  v8f acc[4] = {zero, zero, zero, zero};

  // staging coordinates (constant over the K loop)
  const int wr = tid >> 3;          // W-tile row 0..31
  const int wc = (tid & 7) * 4;     // W-tile k   0,4,..,28
  const int xs = (tid & 31) * 8;    // X-tile s   0,8,..,248
  const int xk = tid >> 5;          // X-tile k (base), rows k = i*8 + xk

  for (int k0 = 0; k0 < C; k0 += 32) {
    __syncthreads();
    // ---- stage W tile: one uint2 (4 bf16) per thread ----
    {
      uint2 q = *(const uint2*)(Wb + (size_t)(oB0 + wr) * C + (k0 + wc));
      *(uint2*)(&Ws[wr * KP + wc]) = q;
    }
    // ---- stage X tile transposed: 4x uint4 (8 bf16) per thread ----
    if (k0 + 32 < C)
      __builtin_prefetch(Xb + (size_t)(k0 + 32 + xk) * S + sB0 + xs, 0, 1);
#pragma unroll
    for (int i = 0; i < 4; ++i) {
      const int k = i * 8 + xk;
      uint4 q = *(const uint4*)(Xb + (size_t)(k0 + k) * S + sB0 + xs);
      unsigned short e[8];
      *(uint4*)e = q;
#pragma unroll
      for (int j = 0; j < 8; ++j) XsT[(xs + j) * KP + k] = e[j];
    }
    __syncthreads();

    // ---- A fragment: 16x32 OxK. Lane half picks K {0..7,16..23} / {8..15,24..31}.
    v16bf a;
    {
      const int rowT = waveO * 16 + l16;
      uint4 w0 = *(const uint4*)(&Ws[rowT * KP + half * 8]);
      uint4 w1 = *(const uint4*)(&Ws[rowT * KP + half * 8 + 16]);
      v8u aw = {w0.x, w0.y, w0.z, w0.w, w1.x, w1.y, w1.z, w1.w};
      a = __builtin_bit_cast(v16bf, aw);
    }
    // ---- B fragments + WMMA. Lane half picks K {0..7,16..23} / {8..15,24..31}.
#pragma unroll
    for (int t = 0; t < 4; ++t) {
      const int colT = waveS * 64 + t * 16 + l16;
      uint4 b0 = *(const uint4*)(&XsT[colT * KP + half * 8]);
      uint4 b1 = *(const uint4*)(&XsT[colT * KP + half * 8 + 16]);
      v8u bw = {b0.x, b0.y, b0.z, b0.w, b1.x, b1.y, b1.z, b1.w};
      v16bf bb = __builtin_bit_cast(v16bf, bw);
      acc[t] = __builtin_amdgcn_wmma_f32_16x16x32_bf16(
          false, a, false, bb, (short)0, acc[t], false, false);
    }
  }

  // ---- epilogue: C/D vgpr r holds (M = r + half*8, N = l16) ----
  float bv[8];
#pragma unroll
  for (int r = 0; r < 8; ++r) bv[r] = bias ? bias[oB0 + waveO * 16 + half * 8 + r] : 0.f;

#pragma unroll
  for (int t = 0; t < 4; ++t) {
    const int col = sB0 + waveS * 64 + t * 16 + l16;
#pragma unroll
    for (int r = 0; r < 8; ++r) {
      const int o = oB0 + waveO * 16 + half * 8 + r;
      float v = acc[t][r] + bv[r];
      if (relu) v = fmaxf(v, 0.f);
      const size_t di = ((size_t)b * O + o) * S + col;
      if (OUTBF) ((unsigned short*)Yv)[di] = f2bfu(v);
      else       ((float*)Yv)[di] = v;
    }
  }
}

// ---------------------------------------------------------------------------
// 2x2 max pool, stride 2.
// ---------------------------------------------------------------------------
__global__ void maxpool_kernel(const float* __restrict__ X, float* __restrict__ Y,
                               int C, int H, int W, int n) {
  const int idx = blockIdx.x * blockDim.x + threadIdx.x;
  if (idx >= n) return;
  const int W2 = W >> 1, H2 = H >> 1;
  const int w2 = idx % W2;
  const int h2 = (idx / W2) % H2;
  const int bc = idx / (W2 * H2);
  const float* p = X + (size_t)bc * H * W + (size_t)(2 * h2) * W + 2 * w2;
  Y[idx] = fmaxf(fmaxf(p[0], p[1]), fmaxf(p[W], p[W + 1]));
}

// ---------------------------------------------------------------------------
// ConvTranspose2d(k=2,s=2) into the first O channels of the concat buffer.
// ---------------------------------------------------------------------------
__global__ void upconv_kernel(const float* __restrict__ X, const float* __restrict__ Wt,
                              const float* __restrict__ bias, float* __restrict__ CAT,
                              int Cin, int O, int H, int W, int catC) {
  __shared__ float w_s[4096];  // Cin*4 <= 4096
  const int o = blockIdx.y;
  const int b = blockIdx.z;
  const int tid = threadIdx.x;
  for (int i = tid; i < Cin * 4; i += blockDim.x)
    w_s[i] = Wt[((size_t)(i >> 2) * O + o) * 4 + (i & 3)];
  __syncthreads();
  const int HW = H * W;
  const float bvv = bias[o];
  const float* Xb = X + (size_t)b * Cin * HW;
  float* out = CAT + ((size_t)b * catC + o) * (size_t)(4 * HW);
  const int W2 = 2 * W;
  for (int hw = blockIdx.x * blockDim.x + tid; hw < HW; hw += gridDim.x * blockDim.x) {
    const int h = hw / W, w = hw % W;
    float a00 = bvv, a01 = bvv, a10 = bvv, a11 = bvv;
    for (int c = 0; c < Cin; ++c) {
      const float xv = Xb[(size_t)c * HW + hw];
      const float* wp = w_s + c * 4;
      a00 += xv * wp[0];
      a01 += xv * wp[1];
      a10 += xv * wp[2];
      a11 += xv * wp[3];
    }
    const int row = 2 * h, col = 2 * w;
    out[(size_t)row * W2 + col]           = a00;
    out[(size_t)row * W2 + col + 1]       = a01;
    out[(size_t)(row + 1) * W2 + col]     = a10;
    out[(size_t)(row + 1) * W2 + col + 1] = a11;
  }
}

__global__ void copy_cat_kernel(const float* __restrict__ E, float* __restrict__ CAT,
                                int C, int HW, int catC, int off) {
  const int b = blockIdx.y;
  const int n = C * HW;
  for (int i = blockIdx.x * blockDim.x + threadIdx.x; i < n; i += gridDim.x * blockDim.x)
    CAT[((size_t)b * catC + off) * HW + i] = E[(size_t)b * n + i];
}

__global__ void final_kernel(const float* __restrict__ X, const float* __restrict__ Wf,
                             const float* __restrict__ bf, float* __restrict__ Y,
                             int C, int HW) {
  __shared__ float w_s[64];
  const int tid = threadIdx.x;
  if (tid < C) w_s[tid] = Wf[tid];
  __syncthreads();
  const int b = blockIdx.y;
  const int hw = blockIdx.x * blockDim.x + tid;
  if (hw >= HW) return;
  float s = bf[0];
  for (int c = 0; c < C; ++c) s += w_s[c] * X[((size_t)b * C + c) * HW + hw];
  Y[(size_t)b * HW + hw] = s;
}

// ---------------------------------------------------------------------------
// Host orchestration. Input order: x, 9 blocks x 11 params, 4 ups x 2, final w/b.
// ---------------------------------------------------------------------------
struct BlkCfg { int cin, cout, H, W, r, pi; };

extern "C" void kernel_launch(void* const* d_in, const int* in_sizes, int n_in,
                              void* d_out, int out_size, void* d_ws, size_t ws_size,
                              hipStream_t stream) {
  (void)in_sizes; (void)n_in; (void)out_size; (void)ws_size;
  const float* x = (const float*)d_in[0];
  float* ws = (float*)d_ws;
  size_t off = 0;
  auto allocf = [&](size_t n) -> float* { float* p = ws + off; off += n; return p; };
  auto alloch = [&](size_t n) -> unsigned short* {  // n halves
    unsigned short* p = (unsigned short*)(ws + off); off += (n + 1) / 2; return p;
  };
  auto P = [&](int i) -> const float* { return (const float*)d_in[i]; };

  // persistent f32 activations
  float* e1   = allocf((size_t)B_N * 64 * 65536);
  float* p1   = allocf((size_t)B_N * 64 * 16384);
  float* e2   = allocf((size_t)B_N * 128 * 16384);
  float* p2   = allocf((size_t)B_N * 128 * 4096);
  float* e3   = allocf((size_t)B_N * 256 * 4096);
  float* p3   = allocf((size_t)B_N * 256 * 1024);
  float* e4   = allocf((size_t)B_N * 512 * 1024);
  float* p4   = allocf((size_t)B_N * 512 * 256);
  float* bn   = allocf((size_t)B_N * 1024 * 256);
  float* cat4 = allocf((size_t)B_N * 1024 * 1024);
  float* d4   = allocf((size_t)B_N * 512 * 1024);
  float* cat3 = allocf((size_t)B_N * 512 * 4096);
  float* d3   = allocf((size_t)B_N * 256 * 4096);
  float* cat2 = allocf((size_t)B_N * 256 * 16384);
  float* d2   = allocf((size_t)B_N * 128 * 16384);
  float* cat1 = allocf((size_t)B_N * 128 * 65536);
  float* d1   = allocf((size_t)B_N * 64 * 65536);
  // reusable scratch
  float*          m1  = allocf(64);
  float*          m2  = allocf(64);
  float*          dwk = allocf((size_t)B_N * 1024 * 9);
  unsigned short* pwk = alloch((size_t)B_N * 512 * 1024);   // bf16, padded [b][o][Cpad]
  unsigned short* ydw = alloch((size_t)B_N * 128 * 65536);  // bf16, [b][Cpad][HW]
  unsigned short* dsc = alloch((size_t)B_N * 64 * 65536);   // bf16, [b][cout][HW]
  unsigned short* wbf = alloch((size_t)1024 * 1024);        // bf16 conv_w

  auto run_block = [&](BlkCfg c, const float* in, float* out) {
    const int HW = c.H * c.W;
    const int cpad = (c.cin % 32) ? ((c.cin + 31) & ~31) : c.cin;
    // dynamic depthwise kernel generator
    meanrelu_gen_kernel<<<dim3(B_N * c.r), 256, 0, stream>>>(
        in, P(c.pi + 0), P(c.pi + 1), m1, c.cin, HW, c.r);
    {
      const int N = c.cin * 9;
      fc_gen_kernel<<<dim3((N + 255) / 256, B_N), 256, 0, stream>>>(
          m1, P(c.pi + 2), P(c.pi + 3), dwk, c.r, N);
    }
    // dynamic pointwise kernel generator (bf16, padded)
    meanrelu_gen_kernel<<<dim3(B_N * c.r), 256, 0, stream>>>(
        in, P(c.pi + 4), P(c.pi + 5), m2, c.cin, HW, c.r);
    {
      const int Ntot = c.cout * cpad;
      fc_gen_bf16_kernel<<<dim3((Ntot + 255) / 256, B_N), 256, 0, stream>>>(
          m2, P(c.pi + 6), P(c.pi + 7), pwk, c.r, c.cin, cpad, Ntot);
    }
    // zero padded ydw channels (enc1 only), then depthwise 3x3 -> bf16
    if (cpad != c.cin) {
      const int n = B_N * cpad * HW;
      zero_bf16_kernel<<<dim3((n + 255) / 256), 256, 0, stream>>>(ydw, n);
    }
    dw3x3_kernel<<<dim3((HW + 255) / 256, B_N * c.cin), 256, 0, stream>>>(
        in, dwk, ydw, c.cin, cpad, c.H, c.W);
    // convert conv_w to bf16
    {
      const int n = c.cout * c.cout;
      cvt_bf16_kernel<<<dim3((n + 255) / 256), 256, 0, stream>>>(P(c.pi + 9), wbf, n);
    }
    // per-sample pointwise GEMM (+dsc_bias, ReLU) -> bf16 dsc   [WMMA]
    wmma_gemm_kernel<true><<<dim3(HW / 256, c.cout / 32, B_N), 256, 0, stream>>>(
        ydw, pwk, P(c.pi + 8), dsc, cpad, c.cout, HW, (long)c.cout * cpad, 1);
    // static 1x1 conv (+bias, ReLU) -> f32 out                  [WMMA]
    wmma_gemm_kernel<false><<<dim3(HW / 256, c.cout / 32, B_N), 256, 0, stream>>>(
        dsc, wbf, P(c.pi + 10), out, c.cout, c.cout, HW, 0L, 1);
  };

  auto pool = [&](const float* in, float* out, int C, int H, int W) {
    const int n = B_N * C * (H / 2) * (W / 2);
    maxpool_kernel<<<dim3((n + 255) / 256), 256, 0, stream>>>(in, out, C, H, W, n);
  };

  auto upcat = [&](const float* in, int pi, const float* skip, float* cat,
                   int Cin, int O, int H, int W) {
    const int HW = H * W;
    upconv_kernel<<<dim3((HW + 255) / 256, O, B_N), 256, 0, stream>>>(
        in, P(pi), P(pi + 1), cat, Cin, O, H, W, 2 * O);
    const int sHW = 4 * HW;
    const int n = O * sHW;
    copy_cat_kernel<<<dim3((n + 255) / 256, B_N), 256, 0, stream>>>(
        skip, cat, O, sHW, 2 * O, O);
  };

  // ---- encoder ----
  run_block({3, 64, 256, 256, 8, 1}, x, e1);
  pool(e1, p1, 64, 256, 256);
  run_block({64, 128, 128, 128, 8, 12}, p1, e2);
  pool(e2, p2, 128, 128, 128);
  run_block({128, 256, 64, 64, 8, 23}, p2, e3);
  pool(e3, p3, 256, 64, 64);
  run_block({256, 512, 32, 32, 8, 34}, p3, e4);
  pool(e4, p4, 512, 32, 32);
  run_block({512, 1024, 16, 16, 16, 45}, p4, bn);

  // ---- decoder ----
  upcat(bn, 56, e4, cat4, 1024, 512, 16, 16);
  run_block({1024, 512, 32, 32, 32, 58}, cat4, d4);
  upcat(d4, 69, e3, cat3, 512, 256, 32, 32);
  run_block({512, 256, 64, 64, 16, 71}, cat3, d3);
  upcat(d3, 82, e2, cat2, 256, 128, 64, 64);
  run_block({256, 128, 128, 128, 8, 84}, cat2, d2);
  upcat(d2, 95, e1, cat1, 128, 64, 128, 128);
  run_block({128, 64, 256, 256, 8, 97}, cat1, d1);

  // ---- final 1x1 conv ----
  final_kernel<<<dim3((65536 + 255) / 256, B_N), 256, 0, stream>>>(
      d1, P(108), P(109), (float*)d_out, 64, 65536);
}
